// AttentionGRUCell_35837207118456
// MI455X (gfx1250) — matrix operand order repaired and checked by reading, not compiled, not run on hardware
//
#include <hip/hip_runtime.h>

// ---------------- CDNA5 bf16 WMMA GEMM with fused GRU epilogues ----------------
// D[128x128 tile] = epilogue( sum_t A_t[128xK_t] * W_t[K_t x 1024] + bias )
// v_wmma_f32_16x16x32_bf16, fp32->bf16 packed convert on the fly, fp32 accum.
// 256 threads = 8 wave32, 2x4 wave grid, 64x32 per wave (4x2 16x16 fragments).
// Double-buffered LDS, software-pipelined global loads, b64-only LDS staging.

#define BM 128
#define BN 128
#define BK 32
#define LDA 40   // padded LDS stride (halves) for A tile rows (128 x 32)
#define LDB 40   // padded LDS stride (halves) for B^T tile rows (128 x 32)

typedef __attribute__((ext_vector_type(16))) __bf16 bf16x16;
typedef __attribute__((ext_vector_type(2)))  __bf16 bf16x2;
typedef __attribute__((ext_vector_type(8)))  unsigned short u16x8;
typedef __attribute__((ext_vector_type(8)))  float f32x8;

__device__ __forceinline__ unsigned int pack2bf(float a, float b) {
  bf16x2 p;
  p.x = (__bf16)a;        // RNE hardware convert
  p.y = (__bf16)b;
  return __builtin_bit_cast(unsigned int, p);
}

__device__ __forceinline__ float sigmoid_(float x) {
  return 1.0f / (1.0f + __expf(-x));
}

// epilogue modes
#define EP_RAW      0   // out = acc + bias
#define EP_SIGMOID  1   // out = sigmoid(acc + bias)
#define EP_SIG_MUL  2   // out = sigmoid(acc + bias) * aux0          (r * h)
#define EP_GRU      3   // st = tanh(acc+bias+aux0(p)); out = (1-aux1)*aux2 + aux1*st
#define EP_RELU     4   // out = max(acc + bias, 0)

template <int MODE, int NBIAS, int NTERMS>
__global__ void __launch_bounds__(256)
gru_wmma_gemm(const float* __restrict__ A0, const float* __restrict__ W0, int K0,
              const float* __restrict__ A1, const float* __restrict__ W1, int K1,
              const float* __restrict__ A2, const float* __restrict__ W2, int K2,
              const float* __restrict__ b0, const float* __restrict__ b1,
              const float* __restrict__ b2,
              const float* __restrict__ aux0, const float* __restrict__ aux1,
              const float* __restrict__ aux2,
              float* __restrict__ out, int N) {
  __shared__ unsigned short Alds[2][BM * LDA];   // [row][k]
  __shared__ unsigned short Blds[2][BN * LDB];   // transposed: [n][k]

  const int tid  = threadIdx.x;
  const int lane = tid & 31;
  const int wave = tid >> 5;
  const int waveM = (wave >> 2) * 64;   // 0 or 64
  const int waveN = (wave & 3) * 32;    // 0,32,64,96
  const int half  = lane >> 4;          // K sub-block select per WMMA layout
  const int l16   = lane & 15;

  const int rowBlk = blockIdx.y * BM;   // batch rows
  const int colBlk = blockIdx.x * BN;   // output cols

  f32x8 acc[4][2];
  #pragma unroll
  for (int tm = 0; tm < 4; ++tm)
    #pragma unroll
    for (int tn = 0; tn < 2; ++tn)
      acc[tm][tn] = (f32x8){0.f,0.f,0.f,0.f,0.f,0.f,0.f,0.f};

  const int totalK = K0 + (NTERMS > 1 ? K1 : 0) + (NTERMS > 2 ? K2 : 0);
  const int nSteps = totalK / BK;

  // per-thread staged tile registers
  float4 aReg[4];        // A: 4 x (4 consecutive k for one row)
  float  bReg[4][4];     // B: 4 x (4 consecutive k for one n)

  // map flat K-step -> (activation ptr, weight ptr, term-K, local k offset)
  auto resolve = [&](int s, const float*& Ap, const float*& Wp, int& K, int& kb) {
    int off = s * BK;
    if (NTERMS == 1 || off < K0) { Ap = A0; Wp = W0; K = K0; kb = off; return; }
    off -= K0;
    if (NTERMS == 2 || off < K1) { Ap = A1; Wp = W1; K = K1; kb = off; return; }
    off -= K1;
    Ap = A2; Wp = W2; K = K2; kb = off;
  };

  auto gload = [&](int s) {
    const float* Ap; const float* Wp; int K, kb;
    resolve(s, Ap, Wp, K, kb);
    #pragma unroll
    for (int i = 0; i < 4; ++i) {
      // A: 1024 float4 slots (128 rows x 8 float4)
      const int sa = tid + i * 256;
      const int r  = sa >> 3;           // row in tile
      const int c4 = sa & 7;            // float4 within 32-K row
      aReg[i] = *(const float4*)(Ap + (size_t)(rowBlk + r) * K + kb + c4 * 4);
      // B: 1024 (n, kgroup) slots; 4 consecutive k for one n
      // per load j: consecutive lanes -> consecutive n (coalesced b32)
      const int n  = sa & 127;
      const int kg = sa >> 7;           // 0..7 -> k = kg*4 .. kg*4+3
      const float* wp = Wp + (size_t)(kb + kg * 4) * N + colBlk + n;
      #pragma unroll
      for (int j = 0; j < 4; ++j) bReg[i][j] = wp[(size_t)j * N];
    }
  };

  auto stage = [&](int buf) {
    #pragma unroll
    for (int i = 0; i < 4; ++i) {
      const int sa = tid + i * 256;
      // A: pack 4 bf16 -> one ds_store_b64 (row-major [r][k])
      const int r  = sa >> 3;
      const int c4 = sa & 7;
      uint2 da;
      da.x = pack2bf(aReg[i].x, aReg[i].y);
      da.y = pack2bf(aReg[i].z, aReg[i].w);
      *(uint2*)(&Alds[buf][r * LDA + c4 * 4]) = da;
      // B: pack 4 bf16 (consecutive k) -> one ds_store_b64 into [n][k]
      const int n  = sa & 127;
      const int kg = sa >> 7;
      uint2 db;
      db.x = pack2bf(bReg[i][0], bReg[i][1]);
      db.y = pack2bf(bReg[i][2], bReg[i][3]);
      *(uint2*)(&Blds[buf][n * LDB + kg * 4]) = db;
    }
  };

  auto compute = [&](int buf) {
    union { bf16x16 v; u16x8 h[2]; } bfrag[2];
    #pragma unroll
    for (int tn = 0; tn < 2; ++tn) {
      // B 32x16 layout: lane = N (l16); half=0 -> K{0..7,16..23}, half=1 -> K{8..15,24..31}
      const unsigned short* p = &Blds[buf][(waveN + tn * 16 + l16) * LDB + half * 8];
      bfrag[tn].h[0] = *(const u16x8*)(p);
      bfrag[tn].h[1] = *(const u16x8*)(p + 16);
    }
    #pragma unroll
    for (int tm = 0; tm < 4; ++tm) {
      // A 16x32 layout: lane = M (l16); half=0 -> K{0..7,16..23}, half=1 -> K{8..15,24..31}
      union { bf16x16 v; u16x8 h[2]; } fa;
      const unsigned short* pa = &Alds[buf][(waveM + tm * 16 + l16) * LDA + half * 8];
      fa.h[0] = *(const u16x8*)(pa);
      fa.h[1] = *(const u16x8*)(pa + 16);
      #pragma unroll
      for (int tn = 0; tn < 2; ++tn) {
        acc[tm][tn] = __builtin_amdgcn_wmma_f32_16x16x32_bf16(
            false, fa.v, false, bfrag[tn].v, (short)0, acc[tm][tn], false, false);
      }
    }
  };

  // -------- software-pipelined main loop (double-buffered LDS) --------
  gload(0);
  int buf = 0;
  for (int s = 0; s < nSteps; ++s) {
    stage(buf);                 // regs(tile s) -> LDS[buf]
    __syncthreads();            // LDS[buf] ready; all waves done with LDS[buf^1]
    if (s + 1 < nSteps) gload(s + 1);   // issue next tile's global loads
    compute(buf);               // 8 WMMAs while loads are in flight
    buf ^= 1;
  }

  // -------- fused epilogue (fully compile-time specialized) --------
  #pragma unroll
  for (int tn = 0; tn < 2; ++tn) {
    const int col = colBlk + waveN + tn * 16 + l16;
    float bias = 0.f;
    if (NBIAS >= 1) bias += b0[col];
    if (NBIAS >= 2) bias += b1[col];
    if (NBIAS >= 3) bias += b2[col];
    #pragma unroll
    for (int tm = 0; tm < 4; ++tm) {
      #pragma unroll
      for (int j = 0; j < 8; ++j) {
        const int row = rowBlk + waveM + tm * 16 + j + half * 8;  // C/D VGPR-j layout
        const size_t idx = (size_t)row * N + col;
        float v = acc[tm][tn][j] + bias;
        if constexpr (MODE == EP_SIGMOID) {
          v = sigmoid_(v);
        } else if constexpr (MODE == EP_SIG_MUL) {
          v = sigmoid_(v) * aux0[idx];
        } else if constexpr (MODE == EP_GRU) {
          const float st = tanhf(v + aux0[idx]);
          const float z  = aux1[idx];
          v = (1.f - z) * aux2[idx] + z * st;
        } else if constexpr (MODE == EP_RELU) {
          v = fmaxf(v, 0.f);
        }
        out[idx] = v;
      }
    }
  }
}

extern "C" void kernel_launch(void* const* d_in, const int* in_sizes, int n_in,
                              void* d_out, int out_size, void* d_ws, size_t ws_size,
                              hipStream_t stream) {
  (void)in_sizes; (void)n_in; (void)out_size; (void)ws_size;
  const int Bt = 16384, N = 1024, EMB = 1024, HIDK = 1024, COMB = 2048;

  const float* X   = (const float*)d_in[0];   // in_word        [B, EMB]
  const float* Hs  = (const float*)d_in[1];   // last_hid_state [B, HID]
  const float* Att = (const float*)d_in[2];   // attended_state [B, COMB]
  const float* W   = (const float*)d_in[3];  const float* bw  = (const float*)d_in[4];
  const float* Wz  = (const float*)d_in[5];  const float* bwz = (const float*)d_in[6];
  const float* Wr  = (const float*)d_in[7];  const float* bwr = (const float*)d_in[8];
  const float* U   = (const float*)d_in[9];  const float* bu  = (const float*)d_in[10];
  const float* Uz  = (const float*)d_in[11]; const float* buz = (const float*)d_in[12];
  const float* Ur  = (const float*)d_in[13]; const float* bur = (const float*)d_in[14];
  const float* C   = (const float*)d_in[15]; const float* bc  = (const float*)d_in[16];
  const float* Cz  = (const float*)d_in[17]; const float* bcz = (const float*)d_in[18];
  const float* Cr  = (const float*)d_in[19]; const float* bcr = (const float*)d_in[20];
  const float* Uo  = (const float*)d_in[21]; const float* buo = (const float*)d_in[22];
  const float* Vo  = (const float*)d_in[23]; const float* bvo = (const float*)d_in[24];
  const float* Co  = (const float*)d_in[25]; const float* bco = (const float*)d_in[26];

  float* ws = (float*)d_ws;
  const size_t plane = (size_t)Bt * N;
  float* zbuf  = ws;               // sigmoid gate z
  float* rhbuf = ws + plane;       // r * h
  float* pbuf  = ws + 2 * plane;   // X@W + A@C partial
  float* s_out = (float*)d_out;            // output s
  float* t_out = s_out + plane;            // output t_i

  dim3 grid(N / BN, Bt / BM);
  dim3 block(256);

  // z = sigmoid(X@Wz + H@Uz + A@Cz + b)
  gru_wmma_gemm<EP_SIGMOID, 3, 3><<<grid, block, 0, stream>>>(
      X, Wz, EMB, Hs, Uz, HIDK, Att, Cz, COMB,
      bwz, buz, bcz, nullptr, nullptr, nullptr, zbuf, N);
  // rh = sigmoid(X@Wr + H@Ur + A@Cr + b) * H
  gru_wmma_gemm<EP_SIG_MUL, 3, 3><<<grid, block, 0, stream>>>(
      X, Wr, EMB, Hs, Ur, HIDK, Att, Cr, COMB,
      bwr, bur, bcr, Hs, nullptr, nullptr, rhbuf, N);
  // p = X@W + A@C + b
  gru_wmma_gemm<EP_RAW, 2, 2><<<grid, block, 0, stream>>>(
      X, W, EMB, Att, C, COMB, nullptr, nullptr, 0,
      bw, bc, nullptr, nullptr, nullptr, nullptr, pbuf, N);
  // s = (1-z)*H + z*tanh(rh@U + bu + p)
  gru_wmma_gemm<EP_GRU, 1, 1><<<grid, block, 0, stream>>>(
      rhbuf, U, HIDK, nullptr, nullptr, 0, nullptr, nullptr, 0,
      bu, nullptr, nullptr, pbuf, zbuf, Hs, s_out, N);
  // t = relu(s@Uo + X@Vo + A@Co + b)
  gru_wmma_gemm<EP_RELU, 3, 3><<<grid, block, 0, stream>>>(
      s_out, Uo, HIDK, X, Vo, EMB, Att, Co, COMB,
      buo, bvo, bco, nullptr, nullptr, nullptr, t_out, N);
}
